// Bi_Dir_ResidueExtractor_35854386987566
// MI455X (gfx1250) — matrix operand order repaired and checked by reading, not compiled, not run on hardware
//
#include <hip/hip_runtime.h>
#include <hip/hip_bf16.h>
#include <math.h>
#include <stdint.h>

typedef __attribute__((ext_vector_type(16))) _Float16 v16h;
typedef __attribute__((ext_vector_type(8)))  float    v8f;

// ---------------------------------------------------------------------
// CDNA5 async global->LDS staging (ASYNCcnt path), guarded so the file
// compiles on toolchains without the gfx1250 async builtins.
// Builtin signature (from hipcc diagnostic): (AS1 int*, AS3 int*, Ii, Ii).
// ---------------------------------------------------------------------
#if defined(__HIP_DEVICE_COMPILE__) && __has_builtin(__builtin_amdgcn_global_load_async_to_lds_b32)
#define HAS_ASYNC_LDS 1
#endif

#ifdef HAS_ASYNC_LDS
typedef __attribute__((address_space(1))) int as1_int;
typedef __attribute__((address_space(3))) int as3_int;
__device__ __forceinline__ void async_g2l_b32(const float* g, float* l)
{
    // LDS flat address truncates to the 32-bit LDS offset (ISA 10.2).
    __builtin_amdgcn_global_load_async_to_lds_b32(
        (as1_int*)(uintptr_t)g, (as3_int*)(uint32_t)(uintptr_t)l, 0, 0);
}
__device__ __forceinline__ void wait_async()
{
#if __has_builtin(__builtin_amdgcn_s_wait_asynccnt)
    __builtin_amdgcn_s_wait_asynccnt(0);
#else
    asm volatile("s_wait_asynccnt 0" ::: "memory");
#endif
}
#endif

// =====================================================================
// Implicit-GEMM 3x3 conv (pad=1) via CDNA5 WMMA.
//   GEMM: M = B*Hout*Wout, N = Cout, K ordered as r*Cin + ci (tap-outer).
//   Block = 128 thr (4 waves) -> 32(M) x 64(N) tile; wave owns a 16-col
//   slice with two f32 accumulators (two v_wmma per K-chunk).
//   When Cin%32==0 a 32-wide K-chunk has uniform tap r => no per-element
//   division anywhere in the hot loop.
// =====================================================================
#define TM 32
#define TN 64

__global__ __launch_bounds__(128)
void conv3x3_wmma_kernel(const float* __restrict__ in, const float* __restrict__ w,
                         const float* __restrict__ bias, float* __restrict__ out,
                         int Bn, int Cin, int Hin, int Win,
                         int Cout, int Hout, int Wout, int stride, int act_silu)
{
    __shared__ float As[TM * 32];   // [m][kk]  f32
    __shared__ float Bs[32 * TN];   // [kk][n]  f32
    const int tid  = threadIdx.x;
    const int lane = tid & 31;
    const int wave = tid >> 5;
    const int HW   = Hout * Wout;
    const int P    = Bn * HW;
    const int Ktot = Cin * 9;
    const int tile_m0 = blockIdx.x * TM;
    const int tile_n0 = blockIdx.y * TN;

    // ---- hoist spatial decode of this thread's A-staging rows (fixed) ----
    int py0[8], px0[8], pb[8]; bool pmv[8];
#pragma unroll
    for (int t = 0; t < 8; ++t) {
        int m  = (tid >> 5) + 4 * t;
        int gm = tile_m0 + m;
        bool v = gm < P;
        int gmc = v ? gm : 0;
        int b  = gmc / HW, rem = gmc - b * HW;
        int oy = rem / Wout, ox = rem - oy * Wout;
        py0[t] = oy * stride - 1;
        px0[t] = ox * stride - 1;
        pb[t]  = b;
        pmv[t] = v;
    }
    const int  kkA  = tid & 31;                 // A staging: K lane (fixed)
    const int  nB   = tid & 63;                 // B staging: column (fixed)
    const int  coB  = tile_n0 + nB;
    const bool coBv = coB < Cout;

    v8f acc0 = {}, acc1 = {};
    const bool fastK = (Cin & 31) == 0;         // chunk never straddles a tap
    int r_cur = 0, ci0 = 0;                     // k0 = r_cur*Cin + ci0

    for (int k0 = 0; k0 < Ktot; k0 += 32) {
        // ---------------- stage A tile (im2col gather) ----------------
        {
            int k  = k0 + kkA;
            int rA  = fastK ? r_cur : (k / Cin);
            int ciA = fastK ? (ci0 + kkA) : (k - rA * Cin);
            int ky  = rA / 3, kx = rA - ky * 3;
            bool kv = k < Ktot;
#pragma unroll
            for (int t = 0; t < 8; ++t) {
                int idx = ((tid >> 5) + 4 * t) * 32 + kkA;
                int iy  = py0[t] + ky;
                int ix  = px0[t] + kx;
                bool v  = kv && pmv[t] &&
                          (unsigned)iy < (unsigned)Hin && (unsigned)ix < (unsigned)Win;
                const float* gp = in + (((long)(pb[t] * Cin + ciA)) * Hin + iy) * (long)Win + ix;
#ifdef HAS_ASYNC_LDS
                if (v) async_g2l_b32(gp, &As[idx]);
                else   As[idx] = 0.f;
#else
                As[idx] = v ? *gp : 0.f;
#endif
            }
        }
        // ---------------- stage B tile (weights) ----------------
        {
#pragma unroll
            for (int t = 0; t < 16; ++t) {
                int kk  = (tid >> 6) + 2 * t;
                int k   = k0 + kk;
                int rB  = fastK ? r_cur : (k / Cin);
                int ciB = fastK ? (ci0 + kk) : (k - rB * Cin);
                bool v  = (k < Ktot) && coBv;
                const float* gp = w + ((long)coB * Cin + ciB) * 9 + rB;
                int idx = kk * TN + nB;
#ifdef HAS_ASYNC_LDS
                if (v) async_g2l_b32(gp, &Bs[idx]);
                else   Bs[idx] = 0.f;
#else
                Bs[idx] = v ? *gp : 0.f;
#endif
            }
        }
#ifdef HAS_ASYNC_LDS
        wait_async();          // this wave's async transfers landed in LDS
#endif
        __syncthreads();       // all waves' staging visible

        // ---- gather fragments (CDNA5 16-bit A(16x32)/B(32x16) layouts) ----
        const int mrow = lane & 15;
        const int hi   = lane >> 4;
        v16h af0, af1, bf;
#pragma unroll
        for (int e = 0; e < 16; ++e) {
            int ka = (e < 8) ? (hi * 8 + e) : (16 + hi * 8 + (e - 8));
            af0[e] = (_Float16)As[mrow * 32 + ka];
            af1[e] = (_Float16)As[(16 + mrow) * 32 + ka];
            bf[e]  = (_Float16)Bs[ka * TN + wave * 16 + mrow];
        }
        acc0 = __builtin_amdgcn_wmma_f32_16x16x32_f16(false, af0, false, bf,
                                                      (short)0, acc0, false, false);
        acc1 = __builtin_amdgcn_wmma_f32_16x16x32_f16(false, af1, false, bf,
                                                      (short)0, acc1, false, false);
        __syncthreads();

        if (fastK) { ci0 += 32; if (ci0 == Cin) { ci0 = 0; ++r_cur; } }
    }

    // ---- epilogue: C layout lanes{0-15}:M=r, lanes{16-31}:M=8+r; N=lane&15
    const int n  = lane & 15;
    const int hi = lane >> 4;
    const int co = tile_n0 + wave * 16 + n;
    float bv = (co < Cout) ? bias[co] : 0.f;
#pragma unroll
    for (int half = 0; half < 2; ++half) {
        v8f a = half ? acc1 : acc0;
#pragma unroll
        for (int r = 0; r < 8; ++r) {
            int gm = tile_m0 + half * 16 + hi * 8 + r;
            if (gm < P && co < Cout) {
                float v = a[r] + bv;
                if (act_silu) v = v / (1.f + __expf(-v));   // SiLU
                int b  = gm / HW, rem = gm - b * HW;
                int oy = rem / Wout, ox = rem - oy * Wout;
                out[(((long)b * Cout + co) * Hout + oy) * (long)Wout + ox] = v;
            }
        }
    }
}

// =====================================================================
// Elementwise / irregular helper kernels (bandwidth-bound; VALU path)
// =====================================================================
__global__ void zero_kernel(float* __restrict__ p, long n)
{
    long i  = (long)blockIdx.x * blockDim.x + threadIdx.x;
    long st = (long)gridDim.x * blockDim.x;
    for (; i < n; i += st) p[i] = 0.f;
}

// Anti-aliased bilinear downsample (triangle kernel scaled by `scale`,
// per-axis normalized, half-pixel centers) + divide by flow factor.
__global__ void resize_flow_kernel(const float* __restrict__ flow, float* __restrict__ out,
                                   int Bn, int Hin, int Win, int res,
                                   float scale, float inv_factor)
{
    int i   = blockIdx.x * blockDim.x + threadIdx.x;
    int tot = Bn * 2 * res * res;
    if (i >= tot) return;
    int x = i % res; int t = i / res;
    int y = t % res; t /= res;
    int c = t % 2;   int b = t / 2;

    float cy = (y + 0.5f) * scale - 0.5f;
    float cx = (x + 0.5f) * scale - 0.5f;
    int y0 = (int)ceilf(cy - scale), y1 = (int)floorf(cy + scale);
    int x0 = (int)ceilf(cx - scale), x1 = (int)floorf(cx + scale);
    const float* img = flow + ((long)(b * 2 + c)) * Hin * (long)Win;

    float accy = 0.f, wsy = 0.f;
    for (int yy = y0; yy <= y1; ++yy) {
        float wy = 1.f - fabsf((float)yy - cy) / scale;
        if (wy <= 0.f || yy < 0 || yy >= Hin) continue;
        float accx = 0.f, wsx = 0.f;
        for (int xx = x0; xx <= x1; ++xx) {
            float wx = 1.f - fabsf((float)xx - cx) / scale;
            if (wx <= 0.f || xx < 0 || xx >= Win) continue;
            accx += wx * img[(long)yy * Win + xx];
            wsx  += wx;
        }
        if (wsx > 0.f) { accy += wy * (accx / wsx); wsy += wy; }
    }
    out[i] = ((wsy > 0.f) ? accy / wsy : 0.f) * inv_factor;
}

// Grouped 3x3 conv on 2-channel flow (groups=2): per-channel 3x3 + bias.
__global__ void conv3x3_g2_kernel(const float* __restrict__ in, const float* __restrict__ w,
                                  const float* __restrict__ bias, float* __restrict__ out,
                                  int Bn, int res)
{
    int i   = blockIdx.x * blockDim.x + threadIdx.x;
    int tot = Bn * 2 * res * res;
    if (i >= tot) return;
    int x = i % res; int t = i / res;
    int y = t % res; t /= res;
    int c = t % 2;   int b = t / 2;
    const float* img = in + ((long)(b * 2 + c)) * res * res;
    const float* wk  = w + c * 9;         // w: (2,1,3,3)
    float s = bias[c];
#pragma unroll
    for (int ky = 0; ky < 3; ++ky) {
#pragma unroll
        for (int kx = 0; kx < 3; ++kx) {
            int iy = y + ky - 1, ix = x + kx - 1;
            if ((unsigned)iy < (unsigned)res && (unsigned)ix < (unsigned)res)
                s += wk[ky * 3 + kx] * img[iy * res + ix];
        }
    }
    out[i] = s;
}

__device__ __forceinline__ float samp0(const float* img, int Hn, int Wn, int xi, int yi)
{
    if ((unsigned)xi >= (unsigned)Wn || (unsigned)yi >= (unsigned)Hn) return 0.f;
    return img[(long)yi * Wn + xi];
}

// fwd/bwd consistency occlusion mask: warp fb by fa, threshold |fa+fb_w|^2.
__global__ void occ_mask_kernel(const float* __restrict__ fa, const float* __restrict__ fb,
                                float* __restrict__ mask, int Bn, int res)
{
    int i   = blockIdx.x * blockDim.x + threadIdx.x;
    int tot = Bn * res * res;
    if (i >= tot) return;
    int x = i % res; int t = i / res;
    int y = t % res; int b = t / res;
    long pp = (long)res * res;
    const float* fab = fa + (long)b * 2 * pp;
    const float* fbb = fb + (long)b * 2 * pp;
    float a0 = fab[(long)y * res + x];
    float a1 = fab[pp + (long)y * res + x];
    float tx = x + a0, ty = y + a1;
    float x0f = floorf(tx), y0f = floorf(ty);
    float fx = tx - x0f, fy = ty - y0f;
    int x0 = (int)x0f, y0 = (int)y0f;
    float w00 = (1.f - fx) * (1.f - fy), w10 = fx * (1.f - fy);
    float w01 = (1.f - fx) * fy,         w11 = fx * fy;
    const float* c0 = fbb;
    const float* c1 = fbb + pp;
    float b0 = samp0(c0, res, res, x0, y0) * w00 + samp0(c0, res, res, x0 + 1, y0) * w10
             + samp0(c0, res, res, x0, y0 + 1) * w01 + samp0(c0, res, res, x0 + 1, y0 + 1) * w11;
    float b1 = samp0(c1, res, res, x0, y0) * w00 + samp0(c1, res, res, x0 + 1, y0) * w10
             + samp0(c1, res, res, x0, y0 + 1) * w01 + samp0(c1, res, res, x0 + 1, y0 + 1) * w11;
    float d0 = a0 + b0, d1 = a1 + b1;
    float diff = d0 * d0 + d1 * d1;
    float mag  = a0 * a0 + a1 * a1 + b0 * b0 + b1 * b1;
    mask[i] = (diff < 0.01f * mag + 0.5f) ? 1.f : 0.f;
}

// 3x3 conv Cin->1, clip(-20,20), exp, multiply by occlusion mask.
__global__ void metric_kernel(const float* __restrict__ feat, const float* __restrict__ w,
                              const float* __restrict__ bias, const float* __restrict__ occ,
                              float* __restrict__ out, int Bn, int C, int res)
{
    int i   = blockIdx.x * blockDim.x + threadIdx.x;
    int tot = Bn * res * res;
    if (i >= tot) return;
    int x = i % res; int t = i / res;
    int y = t % res; int b = t / res;
    long pp = (long)res * res;
    const float* fb = feat + (long)b * C * pp;
    float s = bias[0];
    for (int ci = 0; ci < C; ++ci) {
        const float* img = fb + (long)ci * pp;
        const float* wk  = w + (long)ci * 9;
#pragma unroll
        for (int ky = 0; ky < 3; ++ky) {
#pragma unroll
            for (int kx = 0; kx < 3; ++kx) {
                int iy = y + ky - 1, ix = x + kx - 1;
                if ((unsigned)iy < (unsigned)res && (unsigned)ix < (unsigned)res)
                    s += wk[ky * 3 + kx] * img[(long)iy * res + ix];
            }
        }
    }
    s = fminf(fmaxf(s, -20.f), 20.f);
    out[i] = __expf(s) * occ[i];
}

// Softmax-splat scatter: one thread per (b, c in [0..C], y, x); channel C
// carries the weight itself. 4-corner bilinear scatter via f32 global atomics.
__global__ void splat_kernel(const float* __restrict__ feat, const float* __restrict__ flow,
                             const float* __restrict__ m, float* __restrict__ acc,
                             int Bn, int C, int res)
{
    long i   = (long)blockIdx.x * blockDim.x + threadIdx.x;
    long tot = (long)Bn * (C + 1) * res * res;
    if (i >= tot) return;
    int  x = (int)(i % res); long t = i / res;
    int  y = (int)(t % res); t /= res;
    int  c = (int)(t % (C + 1));
    int  b = (int)(t / (C + 1));
    long pp  = (long)res * res;
    long pix = (long)y * res + x;
    const float* fl = flow + (long)b * 2 * pp;
    float f0 = fl[pix], f1 = fl[pp + pix];
    float wv = m[(long)b * pp + pix];
    float val = (c == C) ? wv : feat[((long)b * C + c) * pp + pix] * wv;
    float tx = x + f0, ty = y + f1;
    float x0f = floorf(tx), y0f = floorf(ty);
    float fx = tx - x0f, fy = ty - y0f;
    int x0 = (int)x0f, y0 = (int)y0f;
    float* dst = acc + ((long)b * (C + 1) + c) * pp;
    float wc[4] = { (1.f - fx) * (1.f - fy), fx * (1.f - fy), (1.f - fx) * fy, fx * fy };
    int   dx[4] = { 0, 1, 0, 1 };
    int   dy[4] = { 0, 0, 1, 1 };
#pragma unroll
    for (int k = 0; k < 4; ++k) {
        int xi = x0 + dx[k], yi = y0 + dy[k];
        if ((unsigned)xi < (unsigned)res && (unsigned)yi < (unsigned)res)
            atomicAdd(&dst[(long)yi * res + xi], val * wc[k]);
    }
}

// warped = acc[:C]/(den+1e-7); conf = max(den,0); fused = sum(conf*warped)/(sum(conf)+1e-6)
__global__ void fuse_kernel(const float* __restrict__ accp, const float* __restrict__ accn,
                            float* __restrict__ fused, int Bn, int C, int res)
{
    long i   = (long)blockIdx.x * blockDim.x + threadIdx.x;
    long tot = (long)Bn * C * res * res;
    if (i >= tot) return;
    int  x = (int)(i % res); long t = i / res;
    int  y = (int)(t % res); t /= res;
    int  c = (int)(t % C);
    int  b = (int)(t / C);
    long pp  = (long)res * res;
    long pix = (long)y * res + x;
    long baseP = (long)b * (C + 1) * pp;
    float dp = accp[baseP + (long)C * pp + pix];
    float dn = accn[baseP + (long)C * pp + pix];
    float wp = accp[baseP + (long)c * pp + pix] / (dp + 1e-7f);
    float wn = accn[baseP + (long)c * pp + pix] / (dn + 1e-7f);
    float cp = fmaxf(dp, 0.f), cn = fmaxf(dn, 0.f);
    fused[((long)b * C + c) * pp + pix] = (cp * wp + cn * wn) / (cp + cn + 1e-6f);
}

// =====================================================================
// Host-side orchestration
// =====================================================================
static void conv_wmma(const float* in, const float* w, const float* b, float* out,
                      int Bn, int Cin, int Hin, int Win, int Cout,
                      int stride, int act, hipStream_t s)
{
    int Hout = (stride == 2) ? Hin / 2 : Hin;
    int Wout = (stride == 2) ? Win / 2 : Win;
    int P = Bn * Hout * Wout;
    dim3 g((P + TM - 1) / TM, (Cout + TN - 1) / TN);
    conv3x3_wmma_kernel<<<g, 128, 0, s>>>(in, w, b, out, Bn, Cin, Hin, Win,
                                          Cout, Hout, Wout, stride, act);
}

extern "C" void kernel_launch(void* const* d_in, const int* in_sizes, int n_in,
                              void* d_out, int out_size, void* d_ws, size_t ws_size,
                              hipStream_t stream)
{
    (void)in_sizes; (void)n_in; (void)out_size; (void)ws_size;
    const float* prev   = (const float*)d_in[0];
    const float* next   = (const float*)d_in[1];
    const float* flow_f = (const float*)d_in[2];
    const float* flow_b = (const float*)d_in[3];
    // params flattened as a sorted-key pytree:
    //  4.. metric(4x wb) | 12.. next_pre(3x) | 18.. next_pyr(4x)
    // 26.. prev_pre(3x)  | 32.. prev_pyr(4x) | 40.. refine(4x) | 48.. zero(4x)
    auto Pp = [&](int i) { return (const float*)d_in[i]; };

    const int ICv[4]  = {320, 640, 1280, 1280};
    const int CSv[4]  = {160, 320, 640, 640};
    const int RESv[4] = {64, 32, 16, 8};
    const int Bn = 4;

    float* ws  = (float*)d_ws;
    float* out = (float*)d_out;

    // ---- workspace layout (float offsets) ----
    float* e1 = ws;                 // 4*32*512*512  = 33,554,432
    float* e2 = ws + 33554432L;     // 4*64*256*256  = 16,777,216
    float* e3 = ws + 50331648L;     // 4*64*128*128  =  4,194,304
    float* pf[4]; float* nf[4];
    long off = 54525952L;
    for (int i = 0; i < 4; ++i) { pf[i] = ws + off; off += (long)Bn * CSv[i] * RESv[i] * RESv[i]; }
    for (int i = 0; i < 4; ++i) { nf[i] = ws + off; off += (long)Bn * CSv[i] * RESv[i] * RESv[i]; }
    // per-level temporaries alias e1 (dead after both encodes)
    float* t_ffr = e1;
    float* t_fbr = e1 + 32768L;
    float* t_ff  = e1 + 65536L;
    float* t_fb  = e1 + 98304L;
    float* t_of  = e1 + 131072L;
    float* t_ob  = e1 + 147456L;
    float* t_mp  = e1 + 163840L;
    float* t_mn  = e1 + 180224L;
    float* t_ap  = e1 + 196608L;    // 4*161*64*64 = 2,637,824 (max level)
    float* t_an  = e1 + 2834432L;
    float* t_fu  = e1 + 5472256L;   // 4*160*64*64 = 2,621,440 (max level)

    // ---- pre_encode + pyramid: prev stream ----
    conv_wmma(prev, Pp(26), Pp(27), e1, Bn,  3, 512, 512, 32, 1, 1, stream);
    conv_wmma(e1,   Pp(28), Pp(29), e2, Bn, 32, 512, 512, 64, 2, 1, stream);
    conv_wmma(e2,   Pp(30), Pp(31), e3, Bn, 64, 256, 256, 64, 2, 1, stream);
    {
        const float* cur = e3; int C = 64, Hh = 128;
        for (int i = 0; i < 4; ++i) {
            conv_wmma(cur, Pp(32 + 2 * i), Pp(33 + 2 * i), pf[i], Bn, C, Hh, Hh, CSv[i], 2, 1, stream);
            cur = pf[i]; C = CSv[i]; Hh /= 2;
        }
    }
    // ---- pre_encode + pyramid: next stream (reuse e1/e2/e3) ----
    conv_wmma(next, Pp(12), Pp(13), e1, Bn,  3, 512, 512, 32, 1, 1, stream);
    conv_wmma(e1,   Pp(14), Pp(15), e2, Bn, 32, 512, 512, 64, 2, 1, stream);
    conv_wmma(e2,   Pp(16), Pp(17), e3, Bn, 64, 256, 256, 64, 2, 1, stream);
    {
        const float* cur = e3; int C = 64, Hh = 128;
        for (int i = 0; i < 4; ++i) {
            conv_wmma(cur, Pp(18 + 2 * i), Pp(19 + 2 * i), nf[i], Bn, C, Hh, Hh, CSv[i], 2, 1, stream);
            cur = nf[i]; C = CSv[i]; Hh /= 2;
        }
    }

    // ---- per-level flow refine, masks, metrics, softsplat, fuse, zero-conv ----
    long ooff[4] = {0L, 5242880L, 7864320L, 9175040L};
    for (int i = 0; i < 4; ++i) {
        const int res = RESv[i];
        const int C   = CSv[i];
        const float scale = 512.f / (float)res;
        const int npix2 = Bn * 2 * res * res;
        const int npix  = Bn * res * res;

        resize_flow_kernel<<<(npix2 + 255) / 256, 256, 0, stream>>>(flow_f, t_ffr, Bn, 512, 512, res, scale, 1.f / scale);
        resize_flow_kernel<<<(npix2 + 255) / 256, 256, 0, stream>>>(flow_b, t_fbr, Bn, 512, 512, res, scale, 1.f / scale);
        conv3x3_g2_kernel<<<(npix2 + 255) / 256, 256, 0, stream>>>(t_ffr, Pp(40 + 2 * i), Pp(41 + 2 * i), t_ff, Bn, res);
        conv3x3_g2_kernel<<<(npix2 + 255) / 256, 256, 0, stream>>>(t_fbr, Pp(40 + 2 * i), Pp(41 + 2 * i), t_fb, Bn, res);
        occ_mask_kernel<<<(npix + 255) / 256, 256, 0, stream>>>(t_ff, t_fb, t_of, Bn, res);
        occ_mask_kernel<<<(npix + 255) / 256, 256, 0, stream>>>(t_fb, t_ff, t_ob, Bn, res);
        metric_kernel<<<(npix + 255) / 256, 256, 0, stream>>>(pf[i], Pp(4 + 2 * i), Pp(5 + 2 * i), t_of, t_mp, Bn, C, res);
        metric_kernel<<<(npix + 255) / 256, 256, 0, stream>>>(nf[i], Pp(4 + 2 * i), Pp(5 + 2 * i), t_ob, t_mn, Bn, C, res);

        long asz = (long)Bn * (C + 1) * res * res;
        zero_kernel<<<1024, 256, 0, stream>>>(t_ap, asz);
        zero_kernel<<<1024, 256, 0, stream>>>(t_an, asz);
        splat_kernel<<<(unsigned)((asz + 255) / 256), 256, 0, stream>>>(pf[i], t_ff, t_mp, t_ap, Bn, C, res);
        splat_kernel<<<(unsigned)((asz + 255) / 256), 256, 0, stream>>>(nf[i], t_fb, t_mn, t_an, Bn, C, res);

        long ftot = (long)Bn * C * res * res;
        fuse_kernel<<<(unsigned)((ftot + 255) / 256), 256, 0, stream>>>(t_ap, t_an, t_fu, Bn, C, res);

        conv_wmma(t_fu, Pp(48 + 2 * i), Pp(49 + 2 * i), out + ooff[i], Bn, C, res, res, ICv[i], 1, 0, stream);
    }
}
